// Attention_50483045598018
// MI455X (gfx1250) — matrix operand order repaired and checked by reading, not compiled, not run on hardware
//
#include <hip/hip_runtime.h>
#include <hip/hip_bf16.h>

// ---------------- problem constants (fixed by reference) ----------------
#define B_DIM   2
#define S_DIM   2048            // 8*16*16
#define H_DIM   16
#define DH_DIM  72
#define C_DIM   1152            // H_DIM * DH_DIM
#define M_DIM   4096            // B_DIM * S_DIM
#define DQK     96              // head dim padded to 3x32 for WMMA K loop
#define DV      80              // head dim padded to 5x16 for WMMA N tiles

typedef __attribute__((ext_vector_type(16))) _Float16 v16h;
typedef __attribute__((ext_vector_type(8)))  _Float16 v8h;
typedef __attribute__((ext_vector_type(8)))  float    v8f;
typedef __attribute__((ext_vector_type(4)))  unsigned int v4u;
typedef __attribute__((ext_vector_type(8)))  int      v8i;
typedef __attribute__((ext_vector_type(4)))  int      v4i;

#if defined(__has_builtin)
#if __has_builtin(__builtin_amdgcn_tensor_load_to_lds) && __has_builtin(__builtin_amdgcn_s_wait_tensorcnt)
#define HAVE_TDM 1
#endif
#if __has_builtin(__builtin_amdgcn_permlane16)
#define HAVE_PERMLANE16 1
#endif
#endif

__device__ __forceinline__ v8f zero8() {
  v8f z;
#pragma unroll
  for (int i = 0; i < 8; ++i) z[i] = 0.0f;
  return z;
}

__device__ __forceinline__ v16h mk16(v8h lo, v8h hi) {
  v16h r;
#pragma unroll
  for (int i = 0; i < 8; ++i) { r[i] = lo[i]; r[i + 8] = hi[i]; }
  return r;
}

__device__ __forceinline__ v8f wmma_f16(v16h a, v16h b, v8f c) {
  // D = A(16x32 f16) * B(32x16 f16) + C(16x16 f32)
  return __builtin_amdgcn_wmma_f32_16x16x32_f16(
      /*neg_a=*/false, a, /*neg_b=*/false, b,
      /*c_mod=*/(short)0, c, /*reuse_a=*/false, /*reuse_b=*/false);
}

// xor-shuffle within each 16-lane half (VALU v_permlane16_b32; no LDS waits)
__device__ __forceinline__ float lxor16(float x, int m, unsigned lo, unsigned hi) {
#if defined(HAVE_PERMLANE16)
  unsigned u = __float_as_uint(x);
  u = __builtin_amdgcn_permlane16(u, u, lo, hi, false, false);
  return __uint_as_float(u);
#else
  return __shfl_xor(x, m, 32);
#endif
}
// permlane16 nibble selectors for lane -> lane^m within a row of 16
#define SEL1_LO 0x67452301u
#define SEL1_HI 0xEFCDAB89u
#define SEL2_LO 0x54761032u
#define SEL2_HI 0xDCFE98BAu
#define SEL4_LO 0x32107654u
#define SEL4_HI 0xBA98FEDCu
#define SEL8_LO 0xFEDCBA98u
#define SEL8_HI 0x76543210u

// ---------------- kernel 1: fp32 -> f16 conversion ----------------
__global__ void cvt_f32_to_f16(const float* __restrict__ src,
                               _Float16* __restrict__ dst, int n) {
  int i = blockIdx.x * blockDim.x + threadIdx.x;
  int stride = gridDim.x * blockDim.x;
  for (; i < n; i += stride) dst[i] = (_Float16)src[i];
}

// ---------------- kernel 2/5: WMMA GEMM  Y = X @ W^T + bias ----------------
// X: [M_DIM x C_DIM] f16 row-major, W: [C_DIM x C_DIM] f16 row-major ([n][k]).
// Block = 128 threads (4 waves). Block tile 64(M) x 64(N); each wave: 64x16.
template <bool OUTF32>
__global__ void __launch_bounds__(128)
gemm_xwT(const _Float16* __restrict__ X,
         const _Float16* __restrict__ W,
         const float* __restrict__ bias,
         void* __restrict__ Yout) {
  __shared__ __align__(16) _Float16 xTile[64 * 32];

  const int tid   = threadIdx.x;
  const int lane  = tid & 31;
  const int wave  = tid >> 5;
  const int mBase = blockIdx.x * 64;
  const int n0    = blockIdx.y * 64 + wave * 16;
  const int nCol  = lane & 15;
  const int hi    = lane >> 4;   // half-wave select
  const int koffA = hi * 8;      // A fragment K sub-offset per ISA layout
  const int koffB = hi * 8;      // B fragment K sub-offset per ISA layout

  v8f acc[4];
#pragma unroll
  for (int i = 0; i < 4; ++i) acc[i] = zero8();

  for (int kk = 0; kk < C_DIM; kk += 32) {
    __syncthreads();
    // cooperative stage of X tile: 64 rows x 32 cols = 256 x 16B
#pragma unroll
    for (int i = 0; i < 2; ++i) {
      int idx = tid * 2 + i;          // 0..255
      int row = idx >> 2;
      int cch = idx & 3;
      *(v8h*)&xTile[row * 32 + cch * 8] =
          *(const v8h*)&X[(size_t)(mBase + row) * C_DIM + kk + cch * 8];
    }
    __syncthreads();

    // B fragment: two 16B chunks per lane (k = 8h.., 16+8h..) from W[n][k]
    const _Float16* bp = &W[(size_t)(n0 + nCol) * C_DIM + kk + koffB];
    v16h bfrag = mk16(*(const v8h*)bp, *(const v8h*)(bp + 16));

    // batch all A-fragment loads, then issue the WMMA chain
    v16h afrag[4];
#pragma unroll
    for (int ms = 0; ms < 4; ++ms) {
      const _Float16* ap = &xTile[(ms * 16 + nCol) * 32 + koffA];
      afrag[ms] = mk16(*(const v8h*)ap, *(const v8h*)(ap + 16));
    }
#pragma unroll
    for (int ms = 0; ms < 4; ++ms) acc[ms] = wmma_f16(afrag[ms], bfrag, acc[ms]);
  }

  const float bvl = bias[n0 + nCol];
#pragma unroll
  for (int ms = 0; ms < 4; ++ms) {
#pragma unroll
    for (int r = 0; r < 8; ++r) {
      int m = mBase + ms * 16 + r + hi * 8;
      size_t o = (size_t)m * C_DIM + n0 + nCol;
      float v = acc[ms][r] + bvl;
      if (OUTF32) ((float*)Yout)[o] = v;
      else        ((_Float16*)Yout)[o] = (_Float16)v;
    }
  }
}

// ---------------- kernel 3: 3D RoPE + head packing ----------------
__global__ void rope_pack(const _Float16* __restrict__ qraw,
                          const _Float16* __restrict__ kraw,
                          const _Float16* __restrict__ vraw,
                          _Float16* __restrict__ q16,
                          _Float16* __restrict__ k16,
                          _Float16* __restrict__ v16T,
                          const int* __restrict__ pF,
                          const int* __restrict__ pH,
                          const int* __restrict__ pW) {
  const int blk = blockIdx.x;          // 0 .. B*H*S-1
  const int s   = blk & (S_DIM - 1);
  const int bh  = blk >> 11;           // S_DIM = 2048 = 1<<11
  const int b   = bh >> 4;
  const int h   = bh & 15;
  const int d   = threadIdx.x;         // 0..95

  const int hh = *pH, ww = *pW;
  (void)pF;

  const size_t inBase  = ((size_t)(b * S_DIM + s)) * C_DIM + h * DH_DIM;
  const size_t outBase = ((size_t)bh * S_DIM + s) * DQK;

  float qv = 0.0f, kv = 0.0f;
  if (d < DH_DIM) {
    const int c = d / 24;
    const int j = d % 24;
    const int t = (j < 12) ? j : j - 12;
    const int pos = (c == 0) ? (s / (hh * ww))
                 : (c == 1) ? ((s / ww) % hh)
                            : (s % ww);
    const float inv = __expf((float)t * (-2.0f / 24.0f) * 9.210340371976184f);
    const float th  = (float)pos * inv;
    const float cs = __cosf(th), sn = __sinf(th);
    const float xq = (float)qraw[inBase + d];
    const float xk = (float)kraw[inBase + d];
    if (j < 12) {
      const float xqp = (float)qraw[inBase + d + 12];
      const float xkp = (float)kraw[inBase + d + 12];
      qv = xq * cs - xqp * sn;
      kv = xk * cs - xkp * sn;
    } else {
      const float xqp = (float)qraw[inBase + d - 12];
      const float xkp = (float)kraw[inBase + d - 12];
      qv = xq * cs + xqp * sn;
      kv = xk * cs + xkp * sn;
    }
    qv *= 0.1178511301977579f;  // 1/sqrt(72) folded into Q
  }
  q16[outBase + d] = (_Float16)qv;
  k16[outBase + d] = (_Float16)kv;

  if (d < DV) {
    float vv = (d < DH_DIM) ? (float)vraw[inBase + d] : 0.0f;
    v16T[((size_t)bh * DV + d) * S_DIM + s] = (_Float16)vv;
  }
}

// ---------------- kernel 4: flash attention (WMMA + TDM staging) ----------
// grid.x = B*H (32), grid.y = S/64 (32), block = 128 (4 waves; 16 Q rows/wave)
__global__ void __launch_bounds__(128, 1)
attn_flash(const _Float16* __restrict__ q16,
           const _Float16* __restrict__ k16,
           const _Float16* __restrict__ v16T,
           _Float16* __restrict__ attn16) {
  __shared__ __align__(16) _Float16 kLds[32 * DQK];   // 6 KB: [32 kv][96]
  __shared__ __align__(16) _Float16 vLds[DV * 32];    // 5 KB: [80 d][32 kv]
  __shared__ __align__(16) _Float16 pLds[4][16 * 32]; // 4 KB: per-wave P tile

  const int tid  = threadIdx.x;
  const int lane = tid & 31;
  const int wave = tid >> 5;
  const int bh   = blockIdx.x;
  const int b    = bh >> 4;
  const int h    = bh & 15;
  const int qBase = blockIdx.y * 64 + wave * 16;
  const int nCol = lane & 15;
  const int hi   = lane >> 4;
  const int koffA = hi * 8;
  const int koffB = hi * 8;

  const _Float16* qhead = q16 + (size_t)bh * S_DIM * DQK;
  const _Float16* khead = k16 + (size_t)bh * S_DIM * DQK;
  const _Float16* vhead = v16T + (size_t)bh * DV * S_DIM;

  // Q fragments register-resident for the whole KV sweep (16 rows x 96)
  v16h qf[3];
#pragma unroll
  for (int kkI = 0; kkI < 3; ++kkI) {
    const _Float16* qp = qhead + (size_t)(qBase + nCol) * DQK + kkI * 32 + koffA;
    qf[kkI] = mk16(*(const v8h*)qp, *(const v8h*)(qp + 16));
  }

  v8f oAcc[5];
#pragma unroll
  for (int i = 0; i < 5; ++i) oAcc[i] = zero8();
  float mrow[8], lrow[8];
#pragma unroll
  for (int r = 0; r < 8; ++r) { mrow[r] = -1e30f; lrow[r] = 0.0f; }

#if defined(HAVE_TDM)
  const unsigned kLdsAddr = (unsigned)(unsigned long long)(&kLds[0]);
  const unsigned vLdsAddr = (unsigned)(unsigned long long)(&vLds[0]);
#endif

  for (int kv0 = 0; kv0 < S_DIM; kv0 += 32) {
    __syncthreads();
#if defined(HAVE_TDM)
    // ---- Tensor Data Mover staging: one DMA pair per block, wave 0 issues.
    if (wave == 0) {
      const v4i gz4 = {0, 0, 0, 0};
      const v8i gz8 = {0, 0, 0, 0, 0, 0, 0, 0};
      {
        // K chunk: contiguous 1-D tile of 32*96 f16 (6 KB)
        unsigned long long ga =
            (unsigned long long)(const void*)(khead + (size_t)kv0 * DQK);
        v4u g0 = {1u, kLdsAddr, (unsigned)ga,
                  (unsigned)(ga >> 32) | 0x80000000u};   // count=1, type=2
        v8i g1 = {(int)0x00010000,          // data_size = 2B
                  (int)(3072u << 16),       // tensor_dim0 = 3072 (lo16 -> [63:48])
                  (int)(1u << 16),          // tensor_dim1 = 1
                  (int)(3072u << 16),       // tile_dim0 = 3072
                  1,                        // tile_dim1 = 1
                  3072,                     // dim0 stride
                  (int)(3072u << 16),       // dim1 stride lo16
                  0};
        __builtin_amdgcn_tensor_load_to_lds(g0, g1, gz4, gz4, gz8, 0);
      }
      {
        // V chunk: 2-D tile, 80 rows x 32 elems, row stride S_DIM
        unsigned long long ga =
            (unsigned long long)(const void*)(vhead + kv0);
        v4u g0 = {1u, vLdsAddr, (unsigned)ga,
                  (unsigned)(ga >> 32) | 0x80000000u};
        v8i g1 = {(int)0x00010000,          // data_size = 2B
                  (int)(2048u << 16),       // tensor_dim0 = S_DIM
                  (int)(80u << 16),         // tensor_dim1 = 80
                  (int)(32u << 16),         // tile_dim0 = 32
                  80,                       // tile_dim1 = 80
                  2048,                     // dim0 stride = S_DIM
                  0, 0};
        __builtin_amdgcn_tensor_load_to_lds(g0, g1, gz4, gz4, gz8, 0);
      }
      __builtin_amdgcn_s_wait_tensorcnt(0);
    }
#else
    // ---- fallback: cooperative vector staging
#pragma unroll
    for (int i = 0; i < 3; ++i) {
      int idx = tid + i * 128;            // 0..383
      int row = idx / 12, cch = idx % 12;
      *(v8h*)&kLds[row * DQK + cch * 8] =
          *(const v8h*)&khead[(size_t)(kv0 + row) * DQK + cch * 8];
    }
#pragma unroll
    for (int i = 0; i < 3; ++i) {
      int idx = tid + i * 128;
      if (idx < 320) {
        int row = idx >> 2, cch = idx & 3;
        *(v8h*)&vLds[row * 32 + cch * 8] =
            *(const v8h*)&vhead[(size_t)row * S_DIM + kv0 + cch * 8];
      }
    }
#endif
    if (kv0 + 32 < S_DIM) {
      __builtin_prefetch(&khead[(size_t)(kv0 + 32 + (tid >> 2)) * DQK], 0, 1);
    }
    __syncthreads();

    // scores: S[16 q x 32 kv]. Batch all 6 K-fragment loads (one DS clause),
    // then run the two independent 3-deep WMMA accumulate chains.
    v16h bk0[3], bk1[3];
#pragma unroll
    for (int kkI = 0; kkI < 3; ++kkI) {
      const _Float16* bp0 = &kLds[(0  + nCol) * DQK + kkI * 32 + koffB];
      const _Float16* bp1 = &kLds[(16 + nCol) * DQK + kkI * 32 + koffB];
      bk0[kkI] = mk16(*(const v8h*)bp0, *(const v8h*)(bp0 + 16));
      bk1[kkI] = mk16(*(const v8h*)bp1, *(const v8h*)(bp1 + 16));
    }
    v8f s0 = zero8(), s1 = zero8();
#pragma unroll
    for (int kkI = 0; kkI < 3; ++kkI) {
      s0 = wmma_f16(qf[kkI], bk0[kkI], s0);
      s1 = wmma_f16(qf[kkI], bk1[kkI], s1);
    }

    // ---- online softmax, staged across rows for ILP (permlane16 = pure VALU)
    float mx[8];
#pragma unroll
    for (int r = 0; r < 8; ++r) mx[r] = fmaxf(s0[r], s1[r]);
#pragma unroll
    for (int r = 0; r < 8; ++r) mx[r] = fmaxf(mx[r], lxor16(mx[r], 1, SEL1_LO, SEL1_HI));
#pragma unroll
    for (int r = 0; r < 8; ++r) mx[r] = fmaxf(mx[r], lxor16(mx[r], 2, SEL2_LO, SEL2_HI));
#pragma unroll
    for (int r = 0; r < 8; ++r) mx[r] = fmaxf(mx[r], lxor16(mx[r], 4, SEL4_LO, SEL4_HI));
#pragma unroll
    for (int r = 0; r < 8; ++r) mx[r] = fmaxf(mx[r], lxor16(mx[r], 8, SEL8_LO, SEL8_HI));

    float corr[8], p0a[8], p1a[8], rs[8];
#pragma unroll
    for (int r = 0; r < 8; ++r) {
      float mnew = fmaxf(mrow[r], mx[r]);
      corr[r] = __expf(mrow[r] - mnew);
      mrow[r] = mnew;
      p0a[r] = __expf(s0[r] - mnew);
      p1a[r] = __expf(s1[r] - mnew);
      rs[r] = p0a[r] + p1a[r];
    }
#pragma unroll
    for (int r = 0; r < 8; ++r) rs[r] += lxor16(rs[r], 1, SEL1_LO, SEL1_HI);
#pragma unroll
    for (int r = 0; r < 8; ++r) rs[r] += lxor16(rs[r], 2, SEL2_LO, SEL2_HI);
#pragma unroll
    for (int r = 0; r < 8; ++r) rs[r] += lxor16(rs[r], 4, SEL4_LO, SEL4_HI);
#pragma unroll
    for (int r = 0; r < 8; ++r) rs[r] += lxor16(rs[r], 8, SEL8_LO, SEL8_HI);
#pragma unroll
    for (int r = 0; r < 8; ++r) {
      lrow[r] = lrow[r] * corr[r] + rs[r];
#pragma unroll
      for (int t5 = 0; t5 < 5; ++t5) oAcc[t5][r] *= corr[r];
      const int prow = r + hi * 8;
      pLds[wave][prow * 32 + nCol]      = (_Float16)p0a[r];
      pLds[wave][prow * 32 + 16 + nCol] = (_Float16)p1a[r];
    }

    // P tile is wave-private: LDS is in-order per wave; fence the compiler only
    __builtin_amdgcn_wave_barrier();
    asm volatile("s_wait_dscnt 0x0" ::: "memory");
    __builtin_amdgcn_wave_barrier();

    // P fragment (16x32 A-layout); batch all 5 V fragments, then WMMA chain
    const _Float16* pp = &pLds[wave][nCol * 32 + koffA];
    v16h pf = mk16(*(const v8h*)pp, *(const v8h*)(pp + 16));
    v16h vf[5];
#pragma unroll
    for (int t5 = 0; t5 < 5; ++t5) {
      const _Float16* vp = &vLds[(t5 * 16 + nCol) * 32 + koffB];
      vf[t5] = mk16(*(const v8h*)vp, *(const v8h*)(vp + 16));
    }
#pragma unroll
    for (int t5 = 0; t5 < 5; ++t5) oAcc[t5] = wmma_f16(pf, vf[t5], oAcc[t5]);
  }

  // epilogue: normalize and write merged-head layout [B,S,C] (f16)
#pragma unroll
  for (int r = 0; r < 8; ++r) {
    const float inv = 1.0f / lrow[r];
    const int srow = qBase + r + hi * 8;
    const size_t ob = (size_t)(b * S_DIM + srow) * C_DIM + h * DH_DIM;
#pragma unroll
    for (int t5 = 0; t5 < 5; ++t5) {
      const int d = t5 * 16 + nCol;
      if (d < DH_DIM) attn16[ob + d] = (_Float16)(oAcc[t5][r] * inv);
    }
  }
}

// ---------------- host-side launch ----------------
extern "C" void kernel_launch(void* const* d_in, const int* in_sizes, int n_in,
                              void* d_out, int out_size, void* d_ws, size_t ws_size,
                              hipStream_t stream) {
  (void)in_sizes; (void)n_in; (void)out_size; (void)ws_size;

  const float* hs = (const float*)d_in[0];
  const float* Wq = (const float*)d_in[1];
  const float* bq = (const float*)d_in[2];
  const float* Wk = (const float*)d_in[3];
  const float* bk = (const float*)d_in[4];
  const float* Wv = (const float*)d_in[5];
  const float* bv = (const float*)d_in[6];
  const float* Wo = (const float*)d_in[7];
  const float* bo = (const float*)d_in[8];
  const int* pF = (const int*)d_in[9];
  const int* pH = (const int*)d_in[10];
  const int* pW = (const int*)d_in[11];

  char* ws = (char*)d_ws;
  size_t off = 0;
  auto take = [&](size_t bytes) {
    char* p = ws + off;
    off += (bytes + 255) & ~(size_t)255;
    return p;
  };
  _Float16* hs16   = (_Float16*)take((size_t)M_DIM * C_DIM * 2);
  _Float16* wq16   = (_Float16*)take((size_t)C_DIM * C_DIM * 2);
  _Float16* wk16   = (_Float16*)take((size_t)C_DIM * C_DIM * 2);
  _Float16* wv16   = (_Float16*)take((size_t)C_DIM * C_DIM * 2);
  _Float16* wo16   = (_Float16*)take((size_t)C_DIM * C_DIM * 2);
  _Float16* qraw16 = (_Float16*)take((size_t)M_DIM * C_DIM * 2);
  _Float16* kraw16 = (_Float16*)take((size_t)M_DIM * C_DIM * 2);
  _Float16* vraw16 = (_Float16*)take((size_t)M_DIM * C_DIM * 2);
  _Float16* q16    = (_Float16*)take((size_t)B_DIM * H_DIM * S_DIM * DQK * 2);
  _Float16* k16    = (_Float16*)take((size_t)B_DIM * H_DIM * S_DIM * DQK * 2);
  _Float16* v16T   = (_Float16*)take((size_t)B_DIM * H_DIM * DV * S_DIM * 2);
  _Float16* attn16 = (_Float16*)take((size_t)M_DIM * C_DIM * 2);

  cvt_f32_to_f16<<<2048, 256, 0, stream>>>(hs, hs16, M_DIM * C_DIM);
  cvt_f32_to_f16<<<1024, 256, 0, stream>>>(Wq, wq16, C_DIM * C_DIM);
  cvt_f32_to_f16<<<1024, 256, 0, stream>>>(Wk, wk16, C_DIM * C_DIM);
  cvt_f32_to_f16<<<1024, 256, 0, stream>>>(Wv, wv16, C_DIM * C_DIM);
  cvt_f32_to_f16<<<1024, 256, 0, stream>>>(Wo, wo16, C_DIM * C_DIM);

  dim3 ggrid(M_DIM / 64, C_DIM / 64);  // 64 x 18
  gemm_xwT<false><<<ggrid, 128, 0, stream>>>(hs16, wq16, bq, (void*)qraw16);
  gemm_xwT<false><<<ggrid, 128, 0, stream>>>(hs16, wk16, bk, (void*)kraw16);
  gemm_xwT<false><<<ggrid, 128, 0, stream>>>(hs16, wv16, bv, (void*)vraw16);

  rope_pack<<<B_DIM * H_DIM * S_DIM, DQK, 0, stream>>>(
      qraw16, kraw16, vraw16, q16, k16, v16T, pF, pH, pW);

  attn_flash<<<dim3(B_DIM * H_DIM, S_DIM / 64), 128, 0, stream>>>(
      q16, k16, v16T, attn16);

  gemm_xwT<true><<<ggrid, 128, 0, stream>>>(attn16, wo16, bo, d_out);
}
